// AttentionHead_15212774162518
// MI455X (gfx1250) — matrix operand order repaired and checked by reading, not compiled, not run on hardware
//
#include <hip/hip_runtime.h>

#define T_SEQ 128
#define C_EMB 64
#define H_DIM 64

typedef __attribute__((ext_vector_type(16))) __bf16    v16bf;
typedef __attribute__((ext_vector_type(2)))  __bf16    v2bf;
typedef __attribute__((ext_vector_type(2)))  float     v2f;
typedef __attribute__((ext_vector_type(8)))  float     v8f;
typedef __attribute__((ext_vector_type(8)))  unsigned  v8u;

// pack two f32 -> two bf16 in one dword (single v_cvt_pk_bf16_f32 via vector fptrunc)
static __device__ __forceinline__ unsigned cvt_pk_bf16(float lo, float hi) {
#if __has_builtin(__builtin_amdgcn_cvt_pk_bf16_f32)
  v2bf p = __builtin_amdgcn_cvt_pk_bf16_f32(lo, hi);
#else
  v2f  f = {lo, hi};
  v2bf p = __builtin_convertvector(f, v2bf);
#endif
  return __builtin_bit_cast(unsigned, p);
}

// single f32 -> bf16 stored straight to LDS (one cvt + one ds_store_b16)
static __device__ __forceinline__ void st_bf16(unsigned short* p, float v) {
  *(__bf16*)p = (__bf16)v;
}

static __device__ __forceinline__ v8f wmma_bf16(v8u a, v8u b, v8f c) {
  // D = A(16x32 bf16) * B(32x16 bf16) + C(16x16 f32)
  return __builtin_amdgcn_wmma_f32_16x16x32_bf16(
      false, __builtin_bit_cast(v16bf, a),
      false, __builtin_bit_cast(v16bf, b),
      (short)0, c, false, false);
}

struct Phase0Buf {
  unsigned short Xbf[T_SEQ * C_EMB];      // 16 KB  x strip, bf16 row-major
  unsigned short Wbf[3][H_DIM * C_EMB];   // 24 KB  Wq,Wk,Wv bf16 row-major [h][c]
};

__global__ __launch_bounds__(256, 1)
void attn_head_kernel(const float* __restrict__ x,
                      const float* __restrict__ Wq,
                      const float* __restrict__ Wk,
                      const float* __restrict__ Wv,
                      float* __restrict__ out) {
  __shared__ union {
    Phase0Buf p0;                          // phase 0/1
    unsigned short P[T_SEQ * T_SEQ];       // 32 KB  phase 2 (probabilities, bf16)
  } u;
  __shared__ unsigned short Qbf[T_SEQ * H_DIM];  // 16 KB
  __shared__ unsigned short Kbf[T_SEQ * H_DIM];  // 16 KB
  __shared__ unsigned short Vt [H_DIM * T_SEQ];  // 16 KB  V transposed: Vt[h][t]

  const int tid  = threadIdx.x;
  const int wave = tid >> 5;       // 0..7 -> row strip [16w, 16w+16)
  const int lane = tid & 31;
  const int ln16 = lane & 15;
  const int half = lane >> 4;      // C/D layout: M = r + 8*half, N = ln16
  const size_t blk = blockIdx.x;

  // ---------- Phase 0: stage x and weights into LDS as bf16 ----------
  {
    const float2* xb = (const float2*)(x + blk * (size_t)(T_SEQ * C_EMB));
    unsigned* Xd = (unsigned*)u.p0.Xbf;
    for (int i = tid; i < T_SEQ * C_EMB / 2; i += 256) {
      float2 f = xb[i];
      Xd[i] = cvt_pk_bf16(f.x, f.y);
    }
    const float2* Ws[3] = {(const float2*)Wq, (const float2*)Wk, (const float2*)Wv};
#pragma unroll
    for (int m = 0; m < 3; ++m) {
      unsigned* Wd = (unsigned*)u.p0.Wbf[m];
      for (int i = tid; i < H_DIM * C_EMB / 2; i += 256) {
        float2 f = Ws[m][i];
        Wd[i] = cvt_pk_bf16(f.x, f.y);
      }
    }
  }
  __syncthreads();

  // ---------- Phase 1: Q,K,V = x @ W^T for this wave's strip ----------
  {
    v8u a0, a1;                                  // A frags (16x32 bf16), both k-steps
    const int row = 16 * wave + ln16;
#pragma unroll
    for (int v = 0; v < 8; ++v) {
      const int k = ((v >> 2) << 4) + (half << 3) + ((v & 3) << 1);
      a0[v] = *(const unsigned*)&u.p0.Xbf[row * C_EMB + k];
      a1[v] = *(const unsigned*)&u.p0.Xbf[row * C_EMB + 32 + k];
    }
#pragma unroll
    for (int mat = 0; mat < 3; ++mat) {
#pragma unroll
      for (int ht = 0; ht < 4; ++ht) {
        // B frag: B[c][h] = W[h][c]; lane n holds row (16ht+n) of W
        const int h = 16 * ht + ln16;
        v8u b0, b1;
#pragma unroll
        for (int v = 0; v < 8; ++v) {
          const int c0 = ((v >> 2) << 4) + (half << 3) + ((v & 3) << 1);
          b0[v] = *(const unsigned*)&u.p0.Wbf[mat][h * C_EMB + c0];
          b1[v] = *(const unsigned*)&u.p0.Wbf[mat][h * C_EMB + 32 + c0];
        }
        v8f acc = {};
        acc = wmma_bf16(a0, b0, acc);
        acc = wmma_bf16(a1, b1, acc);
#pragma unroll
        for (int r = 0; r < 8; r += 2) {
          const int t = 16 * wave + r + (half << 3);       // even
          if (mat == 0) {          // strided rows: direct scalar bf16 stores
            st_bf16(&Qbf[t * H_DIM + h],       acc[r]);
            st_bf16(&Qbf[(t + 1) * H_DIM + h], acc[r + 1]);
          } else if (mat == 1) {
            st_bf16(&Kbf[t * H_DIM + h],       acc[r]);
            st_bf16(&Kbf[(t + 1) * H_DIM + h], acc[r + 1]);
          } else {                 // contiguous pair: one packed cvt + b32 store
            *(unsigned*)&Vt[h * T_SEQ + t] = cvt_pk_bf16(acc[r], acc[r + 1]);
          }
        }
      }
    }
  }
  __syncthreads();

  // ---------- Phase 2: S = QK^T * scale, causal softmax, O = P V ----------
  float S[8][8];      // S[cs][r]: score tile cs for this wave's rows
  {
    v8u qa0, qa1;
    const int row = 16 * wave + ln16;
#pragma unroll
    for (int v = 0; v < 8; ++v) {
      const int k = ((v >> 2) << 4) + (half << 3) + ((v & 3) << 1);
      qa0[v] = *(const unsigned*)&Qbf[row * H_DIM + k];
      qa1[v] = *(const unsigned*)&Qbf[row * H_DIM + 32 + k];
    }
#pragma unroll
    for (int cs = 0; cs < 8; ++cs) {
      if (cs <= wave) {            // wave-uniform: only causal tiles
        v8u b0, b1;                // B[h][s] = K[16cs+s][h]; lane n holds K row 16cs+n
#pragma unroll
        for (int v = 0; v < 8; ++v) {
          const int h0 = ((v >> 2) << 4) + (half << 3) + ((v & 3) << 1);
          b0[v] = *(const unsigned*)&Kbf[(16 * cs + ln16) * H_DIM + h0];
          b1[v] = *(const unsigned*)&Kbf[(16 * cs + ln16) * H_DIM + 32 + h0];
        }
        v8f acc = {};
        acc = wmma_bf16(qa0, b0, acc);
        acc = wmma_bf16(qa1, b1, acc);
#pragma unroll
        for (int r = 0; r < 8; ++r) {
          float s = acc[r] * 0.125f;                  // HEAD_SIZE^-0.5
          const int m = r + (half << 3);
          if (cs == wave && ln16 > m) s = -__builtin_inff();   // causal mask
          S[cs][r] = s;
        }
      } else {
#pragma unroll
        for (int r = 0; r < 8; ++r) S[cs][r] = -__builtin_inff();
      }
    }
  }

  // row softmax: rows live across 16-lane segments (lanes 0-15 / 16-31)
  float mx[8], sm[8];
#pragma unroll
  for (int r = 0; r < 8; ++r) {
    float m = S[0][r];
#pragma unroll
    for (int cs = 1; cs < 8; ++cs) m = fmaxf(m, S[cs][r]);
    mx[r] = m;
  }
#pragma unroll
  for (int off = 1; off < 16; off <<= 1)
#pragma unroll
    for (int r = 0; r < 8; ++r)
      mx[r] = fmaxf(mx[r], __shfl_xor(mx[r], off, 16));
#pragma unroll
  for (int r = 0; r < 8; ++r) sm[r] = 0.f;
#pragma unroll
  for (int cs = 0; cs < 8; ++cs) {
    if (cs <= wave) {              // wave-uniform: masked tiles contribute exactly 0
#pragma unroll
      for (int r = 0; r < 8; ++r) {
        float e = __expf(S[cs][r] - mx[r]);   // diagonal -inf entries -> 0
        S[cs][r] = e;
        sm[r] += e;
      }
    } else {
#pragma unroll
      for (int r = 0; r < 8; ++r) S[cs][r] = 0.f;
    }
  }
#pragma unroll
  for (int off = 1; off < 16; off <<= 1)
#pragma unroll
    for (int r = 0; r < 8; ++r)
      sm[r] += __shfl_xor(sm[r], off, 16);

  // write P (bf16) into this wave's own strip; masked tiles store 0
  float inv[8];
#pragma unroll
  for (int r = 0; r < 8; ++r) inv[r] = 1.0f / sm[r];
#pragma unroll
  for (int r = 0; r < 8; ++r) {
    const int t = 16 * wave + r + (half << 3);
#pragma unroll
    for (int cs = 0; cs < 8; ++cs)
      st_bf16(&u.P[t * T_SEQ + 16 * cs + ln16], S[cs][r] * inv[r]);
  }
  // no cross-wave barrier needed: each wave reads only its own P strip

  // O = P @ V : A = own P rows; only k-steps covering the causal span are nonzero
  const int nks = (wave >> 1) + 1;           // ceil(16*(wave+1)/32)
  v8u pa[4];
#pragma unroll
  for (int ks = 0; ks < 4; ++ks) {
    if (ks < nks) {
      const int prow = 16 * wave + ln16;
#pragma unroll
      for (int v = 0; v < 8; ++v) {
        const int ka = 32 * ks + ((v >> 2) << 4) + (half << 3) + ((v & 3) << 1);
        pa[ks][v] = *(const unsigned*)&u.P[prow * T_SEQ + ka];
      }
    }
  }
#pragma unroll
  for (int ht = 0; ht < 4; ++ht) {
    v8f acc = {};
#pragma unroll
    for (int ks = 0; ks < 4; ++ks) {
      if (ks < nks) {              // wave-uniform: skip all-zero P k-steps
        v8u b;                     // B[s][h] = V[s][h]; lane n holds Vt row 16ht+n
#pragma unroll
        for (int v = 0; v < 8; ++v) {
          const int kb = 32 * ks + ((v >> 2) << 4) + (half << 3) + ((v & 3) << 1);
          b[v] = *(const unsigned*)&Vt[(16 * ht + ln16) * T_SEQ + kb];
        }
        acc = wmma_bf16(pa[ks], b, acc);
      }
    }
#pragma unroll
    for (int r = 0; r < 8; ++r) {
      const int t = 16 * wave + r + (half << 3);
      out[(blk * T_SEQ + t) * (size_t)H_DIM + 16 * ht + ln16] = acc[r];
    }
  }
}

extern "C" void kernel_launch(void* const* d_in, const int* in_sizes, int n_in,
                              void* d_out, int out_size, void* d_ws, size_t ws_size,
                              hipStream_t stream) {
  (void)n_in; (void)d_ws; (void)ws_size; (void)out_size;
  const float* x  = (const float*)d_in[0];
  const float* Wq = (const float*)d_in[1];
  const float* Wk = (const float*)d_in[2];
  const float* Wv = (const float*)d_in[3];
  float* out = (float*)d_out;
  const int Bn = in_sizes[0] / (T_SEQ * C_EMB);   // 4096 batches
  attn_head_kernel<<<dim3(Bn), dim3(256), 0, stream>>>(x, Wq, Wk, Wv, out);
}
